// DecoderLayer_50534585205086
// MI455X (gfx1250) — compile-verified, matrix operand order checked by reading, not run on hardware
//
#include <hip/hip_runtime.h>
#include <hip/hip_bf16.h>

// ---------------- problem constants ----------------
constexpr int CB  = 4;
constexpr int CS  = 4096;
constexpr int CDM = 1024;
constexpr int CH  = 16;
constexpr int CDF = 4096;
constexpr int CDD = 64;      // head dim
constexpr int CKER = 4;
constexpr int CKW  = 3;
constexpr int CL  = CS / CKER;          // 1024 pooled length
constexpr float CNORM = 0.35355339059327373f; // 64^-0.25
constexpr float CNEG  = -1.0e9f;

typedef _Float16 f16;
typedef __attribute__((ext_vector_type(16))) _Float16 v16h;
typedef __attribute__((ext_vector_type(8)))  _Float16 v8h;
typedef __attribute__((ext_vector_type(8)))  float    v8f;
typedef __attribute__((ext_vector_type(4)))  float    v4f;
typedef __attribute__((ext_vector_type(4)))  unsigned int u32x4;
typedef __attribute__((ext_vector_type(8)))  unsigned int u32x8;

// ---------------- WMMA helpers ----------------
__device__ __forceinline__ v8f wmma_f16(v16h a, v16h b, v8f c) {
  // D = A(16x32,f16) * B(32x16,f16) + C(16x16,f32)
  return __builtin_amdgcn_wmma_f32_16x16x32_f16(
      false, a, false, b, (short)0, c, false, false);
}

// Load one lane's A/B fragment row: 16 f16 at k-offsets {half*8..+8} and {16+half*8..+8}
__device__ __forceinline__ v16h load_frag_row(const f16* rowp, int half) {
  v8h lo = *(const v8h*)(rowp + half * 8);
  v8h hi = *(const v8h*)(rowp + 16 + half * 8);
  v16h r;
#pragma unroll
  for (int i = 0; i < 8; ++i) { r[i] = lo[i]; r[i + 8] = hi[i]; }
  return r;
}

__device__ __forceinline__ v16h load_frag(const f16* p, int stride) {
  int lane = threadIdx.x & 31;
  return load_frag_row(p + (size_t)(lane & 15) * stride, lane >> 4);
}

// ---------------- Tensor Data Mover: 2-D tile (rows x 32 f16) global -> LDS ----------------
// Loads `rows` rows of 32 f16 from a row-major matrix with row stride K (f16 units),
// depositing into LDS with an 8-f16 pad after each row (LDS row stride = 40 f16 = 80B),
// which both matches the WMMA fragment loader and kills LDS bank conflicts.
// D# packing per CDNA5 ISA 8.3/8.4 (2-D: groups 2/3 omitted -> 2-SGPR-group form).
__device__ __forceinline__ void tdm_load_tile(unsigned lds_off, const f16* gptr,
                                              unsigned K, unsigned rows) {
  unsigned long long ga = (unsigned long long)(uintptr_t)gptr;
  u32x4 g0;
  g0[0] = 1u;                                   // count=1 (valid user descriptor)
  g0[1] = lds_off;                              // LDS byte address
  g0[2] = (unsigned)ga;                         // global_addr[31:0]
  g0[3] = (unsigned)((ga >> 32) & 0x1FFFFFFu)   // global_addr[56:32]
        | (2u << 30);                           // type=2 ("image")
  u32x8 g1;
  g1[0] = (1u << 16)                            // data_size = 2 bytes (f16)
        | (1u << 20)                            // pad_enable
        | (3u << 22)                            // pad_interval: 16 DWORDs (=64B row)
        | (3u << 25);                           // pad_amount: 4 DWORDs (=8 f16)
  g1[1] = (K & 0xFFFFu) << 16;                  // tensor_dim0[15:0]
  g1[2] = (K >> 16) | (0u << 16);               // tensor_dim0[31:16] | tensor_dim1[15:0]
  g1[3] = 0x10u | (32u << 16);                  // tensor_dim1[31:16] (dim1=1<<20) | tile_dim0=32
  g1[4] = rows;                                 // tile_dim1 | tile_dim2=0
  g1[5] = K;                                    // tensor_dim0_stride[31:0]
  g1[6] = 0u;                                   // stride0[47:32] | stride1[15:0]
  g1[7] = 0u;                                   // stride1[47:16]
  asm volatile("tensor_load_to_lds %0, %1" :: "s"(g0), "s"(g1) : "memory");
}

// ---------------- weight prep: f32 [R,C] -> f16 [C,R] ----------------
__global__ __launch_bounds__(256) void transpose_f32_f16(
    const float* __restrict__ src, f16* __restrict__ dst, int R, int C) {
  __shared__ float tile[32][33];
  int c0 = blockIdx.x * 32, r0 = blockIdx.y * 32;
  int tx = threadIdx.x & 31, ty = threadIdx.x >> 5;   // 32 x 8
#pragma unroll
  for (int i = ty; i < 32; i += 8)
    tile[i][tx] = src[(size_t)(r0 + i) * C + c0 + tx];
  __syncthreads();
#pragma unroll
  for (int i = ty; i < 32; i += 8)
    dst[(size_t)(c0 + i) * R + r0 + tx] = (f16)tile[tx][i];
}

// ---------------- x_embed = x_enc + x_pos (f16 copy for GEMM A) ----------------
__global__ __launch_bounds__(256) void add_embed(
    const float* __restrict__ a, const float* __restrict__ b,
    f16* __restrict__ o16) {
  size_t i = (size_t)blockIdx.x * 256 + threadIdx.x;
  o16[i] = (f16)(a[i] + b[i]);
}

// ---------------- tiled WMMA GEMM with TDM double-buffered staging ----------------
// C[M,N] = act((A@B^T + bias) * scale)
// A: f16 [M,K] row-major ; Bt: f16 [N,K] row-major ; bias: f32 [N]
// act: 0 = identity, 1 = relu^2 ; writes f32 and/or f16 (null = skip)
#define BM 128
#define BN 128
#define BK 32
#define LSTR 40   // LDS row stride in f16 (32 data + 8 TDM pad), 80B keeps 16B alignment

__global__ __launch_bounds__(256) void gemm_wmma(
    const f16* __restrict__ A, const f16* __restrict__ Bt,
    const float* __restrict__ bias,
    float* __restrict__ Cf32, f16* __restrict__ Cf16,
    int M, int N, int K, float scale, int act) {
  __shared__ f16 as[2][BM * LSTR];
  __shared__ f16 bs[2][BN * LSTR];
  const int bm = blockIdx.x * BM;
  const int bn = blockIdx.y * BN;
  const int tid = threadIdx.x;
  const int lane = tid & 31, wave = tid >> 5;
  const int wr = wave >> 2, wcn = wave & 3;   // 2x4 wave grid, each wave 64x32
  const int nlo = lane & 15, half = lane >> 4;

  const f16* Abase = A  + (size_t)bm * K;
  const f16* Bbase = Bt + (size_t)bn * K;
  const unsigned ldsA0 = (unsigned)(uintptr_t)&as[0][0];
  const unsigned ldsA1 = (unsigned)(uintptr_t)&as[1][0];
  const unsigned ldsB0 = (unsigned)(uintptr_t)&bs[0][0];
  const unsigned ldsB1 = (unsigned)(uintptr_t)&bs[1][0];

  v8f acc[4][2];
#pragma unroll
  for (int mt = 0; mt < 4; ++mt)
#pragma unroll
    for (int nt = 0; nt < 2; ++nt)
#pragma unroll
      for (int r = 0; r < 8; ++r) acc[mt][nt][r] = 0.0f;

  const int nk = K / BK;
  // prologue: wave 0 kicks off DMA of the first A/B tiles
  if (wave == 0) {
    tdm_load_tile(ldsA0, Abase, (unsigned)K, BM);
    tdm_load_tile(ldsB0, Bbase, (unsigned)K, BN);
  }

  for (int kt = 0; kt < nk; ++kt) {
    const int cur = kt & 1;
    if (wave == 0) {
      if (kt + 1 < nk) {
        // issue next tiles into the other buffer, then wait for current pair
        tdm_load_tile(cur ? ldsA0 : ldsA1, Abase + (size_t)(kt + 1) * BK,
                      (unsigned)K, BM);
        tdm_load_tile(cur ? ldsB0 : ldsB1, Bbase + (size_t)(kt + 1) * BK,
                      (unsigned)K, BN);
        __builtin_amdgcn_s_wait_tensorcnt(2);   // oldest 2 (current tiles) done
      } else {
        __builtin_amdgcn_s_wait_tensorcnt(0);
      }
    }
    __syncthreads();   // current buffer visible to all 8 waves

    const f16* ac = &as[cur][0];
    const f16* bc = &bs[cur][0];
    // Issue all 12 ds_load_b128 first, then fire 8 independent WMMAs
    // back-to-back (XDL-friendly; ISA 7.5).
    v16h bf[2], af[4];
#pragma unroll
    for (int nt = 0; nt < 2; ++nt)
      bf[nt] = load_frag(bc + (wcn * 32 + nt * 16) * LSTR, LSTR);
#pragma unroll
    for (int mt = 0; mt < 4; ++mt)
      af[mt] = load_frag(ac + (wr * 64 + mt * 16) * LSTR, LSTR);
#pragma unroll
    for (int mt = 0; mt < 4; ++mt) {
      acc[mt][0] = wmma_f16(af[mt], bf[0], acc[mt][0]);
      acc[mt][1] = wmma_f16(af[mt], bf[1], acc[mt][1]);
    }
    __syncthreads();   // all waves done with this buffer before DMA overwrites it
  }

  // epilogue
#pragma unroll
  for (int mt = 0; mt < 4; ++mt) {
#pragma unroll
    for (int nt = 0; nt < 2; ++nt) {
      int col = bn + wcn * 32 + nt * 16 + nlo;
      float bia = bias ? bias[col] : 0.0f;
#pragma unroll
      for (int r = 0; r < 8; ++r) {
        int row = bm + wr * 64 + mt * 16 + half * 8 + r;
        float v = (acc[mt][nt][r] + bia) * scale;
        if (act == 1) { v = fmaxf(v, 0.0f); v = v * v; }
        size_t idx = (size_t)row * N + col;
        if (Cf32) Cf32[idx] = v;
        if (Cf16) Cf16[idx] = (f16)v;
      }
    }
  }
}

// ---------------- causal depthwise conv + head split ----------------
// x: f32 [B,S,DM], w: f32 [KW,DM], b: f32[DM]  -> out f16 [B,H,S,DD]
__global__ __launch_bounds__(256) void dwconv_split(
    const float* __restrict__ x, const float* __restrict__ w,
    const float* __restrict__ b, f16* __restrict__ out) {
  size_t i = (size_t)blockIdx.x * 256 + threadIdx.x;   // over B*S*DM
  int c = (int)(i % CDM);
  size_t bs = i / CDM;
  int s  = (int)(bs % CS);
  int bb = (int)(bs / CS);
  float acc = b[c];
#pragma unroll
  for (int j = 0; j < CKW; ++j) {
    int sj = s - (CKW - 1) + j;
    if (sj >= 0) acc += x[((size_t)bb * CS + sj) * CDM + c] * w[j * CDM + c];
  }
  int h = c / CDD, d = c % CDD;
  out[(((size_t)bb * CH + h) * CS + s) * CDD + d] = (f16)acc;
}

// ---------------- causal avg-pool over seq (KER=4) ----------------
// in: f16 [B,H,S,DD] -> out f16 [B,H,L,DD] (transposed=0) or [B,H,DD,L] (transposed=1)
__global__ __launch_bounds__(256) void avgpool_seq(
    const f16* __restrict__ in, f16* __restrict__ out, int transposed) {
  size_t i = (size_t)blockIdx.x * 256 + threadIdx.x;   // over B*H*L*DD
  int d = (int)(i % CDD);
  size_t t = i / CDD;
  int l = (int)(t % CL);
  size_t bh = t / CL;
  float acc = 0.0f;
#pragma unroll
  for (int j = 0; j < CKER; ++j) {
    int s = l * CKER - (CKER - 1) + j;
    if (s >= 0) acc += (float)in[(bh * CS + s) * CDD + d];
  }
  acc *= 0.25f;
  if (!transposed) out[(bh * CL + l) * CDD + d] = (f16)acc;
  else             out[(bh * CDD + d) * CL + l] = (f16)acc;
}

// ---------------- pooled causal flash attention ----------------
// qp,kp: f16 [B,H,L,DD] ; vpT: f16 [B,H,DD,L] ; o: f16 [B,H,L,DD]
// q,k already scaled by DD^-0.25 each.
__global__ __launch_bounds__(256) void attn_pooled(
    const f16* __restrict__ qp, const f16* __restrict__ kp,
    const f16* __restrict__ vpT, f16* __restrict__ o) {
  __shared__ f16 psh[8 * 16 * LSTR];    // per-wave P scratch (16 x 32, stride 40)
  const int bh   = blockIdx.y;
  const int wave = threadIdx.x >> 5, lane = threadIdx.x & 31;
  const int nlo  = lane & 15, half = lane >> 4;
  const int qr0  = blockIdx.x * 128 + wave * 16;
  f16* pw = psh + wave * 16 * LSTR;

  // Q fragments for this wave's 16 rows (constant over key loop)
  const f16* qrow = qp + ((size_t)bh * CL + qr0 + nlo) * CDD;
  v16h aq0 = load_frag_row(qrow,      half);
  v16h aq1 = load_frag_row(qrow + 32, half);

  float mrow[8], lrow[8];
  v8f oacc[4];
#pragma unroll
  for (int r = 0; r < 8; ++r) { mrow[r] = -3.0e38f; lrow[r] = 0.0f; }
#pragma unroll
  for (int nt = 0; nt < 4; ++nt)
#pragma unroll
    for (int r = 0; r < 8; ++r) oacc[nt][r] = 0.0f;

  const int kend = qr0 + 15;            // causal upper bound for this wave
  for (int kb = 0; kb <= kend; kb += 32) {
    // S = Q * K^T  (two 16-key tiles, contraction over d in two chunks)
    v8f st[2];
#pragma unroll
    for (int t = 0; t < 2; ++t) {
#pragma unroll
      for (int r = 0; r < 8; ++r) st[t][r] = 0.0f;
      const f16* krow = kp + ((size_t)bh * CL + kb + t * 16 + nlo) * CDD;
      v16h bk0 = load_frag_row(krow,      half);
      v16h bk1 = load_frag_row(krow + 32, half);
      st[t] = wmma_f16(aq0, bk0, st[t]);
      st[t] = wmma_f16(aq1, bk1, st[t]);
    }

    // online softmax (rows live across 16-lane groups; reduce with width-16 xor)
    float corr[8];
#pragma unroll
    for (int r = 0; r < 8; ++r) {
      int qrw = qr0 + half * 8 + r;
      float s0 = st[0][r] + ((kb + nlo)      > qrw ? CNEG : 0.0f);
      float s1 = st[1][r] + ((kb + 16 + nlo) > qrw ? CNEG : 0.0f);
      float mx = fmaxf(s0, s1);
#pragma unroll
      for (int msk = 8; msk >= 1; msk >>= 1)
        mx = fmaxf(mx, __shfl_xor(mx, msk, 16));
      float mnew = fmaxf(mrow[r], mx);
      float c  = __expf(mrow[r] - mnew);
      float p0 = __expf(s0 - mnew);
      float p1 = __expf(s1 - mnew);
      float rs = p0 + p1;
#pragma unroll
      for (int msk = 8; msk >= 1; msk >>= 1)
        rs += __shfl_xor(rs, msk, 16);
      lrow[r] = lrow[r] * c + rs;
      mrow[r] = mnew;
      corr[r] = c;
      // C-layout -> A-layout conversion of P via per-wave LDS scratch
      pw[(half * 8 + r) * LSTR + nlo]      = (f16)p0;
      pw[(half * 8 + r) * LSTR + 16 + nlo] = (f16)p1;
    }
#pragma unroll
    for (int nt = 0; nt < 4; ++nt)
#pragma unroll
      for (int r = 0; r < 8; ++r) oacc[nt][r] *= corr[r];

    // DS ops are wave-in-order; fence stops compiler reordering load before store
    asm volatile("s_wait_dscnt 0" ::: "memory");

    v16h pf = load_frag_row(pw + nlo * LSTR, half);
#pragma unroll
    for (int nt = 0; nt < 4; ++nt) {
      const f16* vrow = vpT + ((size_t)bh * CDD + nt * 16 + nlo) * CL + kb;
      v16h bv = load_frag_row(vrow, half);
      oacc[nt] = wmma_f16(pf, bv, oacc[nt]);
    }
  }

  // normalize and store
  float inv[8];
#pragma unroll
  for (int r = 0; r < 8; ++r) inv[r] = 1.0f / lrow[r];
#pragma unroll
  for (int nt = 0; nt < 4; ++nt)
#pragma unroll
    for (int r = 0; r < 8; ++r) {
      int lq = qr0 + half * 8 + r;
      int d  = nt * 16 + nlo;
      o[((size_t)bh * CL + lq) * CDD + d] = (f16)(oacc[nt][r] * inv[r]);
    }
}

// ---------------- nearest upsample (repeat KER) + per-head wup GEMM + head merge ----------------
// o: f16 [B,H,L,DD] ; wupT: f16 [DD,DD] (row d', col d) ; out: f16 [B,S,DM]
__global__ __launch_bounds__(256) void upsample_wup(
    const f16* __restrict__ o, const f16* __restrict__ wupT,
    const float* __restrict__ bup, f16* __restrict__ out) {
  const int bh = blockIdx.y;
  const int bb = bh / CH, h = bh % CH;
  const int wave = threadIdx.x >> 5, lane = threadIdx.x & 31;
  const int nlo = lane & 15, half = lane >> 4;
  const int s  = blockIdx.x * 128 + wave * 16 + nlo;   // A-frag row for this lane
  const int lp = s >> 2;                               // repeat: pooled row
  const f16* ob = o + ((size_t)bh * CL + lp) * CDD;
  v16h a0 = load_frag_row(ob,      half);
  v16h a1 = load_frag_row(ob + 32, half);

  v8f acc[4];
#pragma unroll
  for (int nt = 0; nt < 4; ++nt) {
#pragma unroll
    for (int r = 0; r < 8; ++r) acc[nt][r] = 0.0f;
    const f16* wb = wupT + (size_t)(nt * 16 + nlo) * CDD;
    acc[nt] = wmma_f16(a0, load_frag_row(wb,      half), acc[nt]);
    acc[nt] = wmma_f16(a1, load_frag_row(wb + 32, half), acc[nt]);
  }
#pragma unroll
  for (int nt = 0; nt < 4; ++nt)
#pragma unroll
    for (int r = 0; r < 8; ++r) {
      int srow = blockIdx.x * 128 + wave * 16 + half * 8 + r;
      int dp = nt * 16 + nlo;
      out[((size_t)bb * CS + srow) * CDM + h * CDD + dp] =
          (f16)(acc[nt][r] + bup[dp]);
    }
}

// ---------------- residual + layernorm (one row of DM=1024 per block) ----------------
// out = resA (+resB) + LN(x; g, be) ; writes f32 and optional f16
__global__ __launch_bounds__(256) void residual_ln(
    const float* __restrict__ resA, const float* __restrict__ resB,
    const float* __restrict__ x, const float* __restrict__ g,
    const float* __restrict__ be, float* __restrict__ out32,
    f16* __restrict__ out16) {
  __shared__ float rs[8], rss[8];
  const size_t row = blockIdx.x;
  const int tid = threadIdx.x, lane = tid & 31, wave = tid >> 5;
  const float* xr = x + row * CDM;
  v4f v = *(const v4f*)(xr + tid * 4);
  float s = v[0] + v[1] + v[2] + v[3];
  float ss = v[0]*v[0] + v[1]*v[1] + v[2]*v[2] + v[3]*v[3];
#pragma unroll
  for (int m = 16; m >= 1; m >>= 1) {
    s  += __shfl_xor(s,  m, 32);
    ss += __shfl_xor(ss, m, 32);
  }
  if (lane == 0) { rs[wave] = s; rss[wave] = ss; }
  __syncthreads();
  float ts = 0.0f, tss = 0.0f;
#pragma unroll
  for (int i = 0; i < 8; ++i) { ts += rs[i]; tss += rss[i]; }
  const float mean = ts * (1.0f / CDM);
  const float var  = tss * (1.0f / CDM) - mean * mean;
  const float rstd = rsqrtf(var + 1e-6f);
#pragma unroll
  for (int j = 0; j < 4; ++j) {
    int c = tid * 4 + j;
    float y = (v[j] - mean) * rstd * g[c] + be[c];
    float r = resA[row * CDM + c] + (resB ? resB[row * CDM + c] : 0.0f) + y;
    out32[row * CDM + c] = r;
    if (out16) out16[row * CDM + c] = (f16)r;
  }
}

// ---------------- host side ----------------
extern "C" void kernel_launch(void* const* d_in, const int* in_sizes, int n_in,
                              void* d_out, int out_size, void* d_ws, size_t ws_size,
                              hipStream_t stream) {
  const float* x_enc = (const float*)d_in[0];
  const float* x_pos = (const float*)d_in[1];
  const float* wq  = (const float*)d_in[2];
  const float* bq  = (const float*)d_in[3];
  const float* wk  = (const float*)d_in[4];
  const float* bk  = (const float*)d_in[5];
  const float* wv  = (const float*)d_in[6];
  const float* bv  = (const float*)d_in[7];
  const float* dwq = (const float*)d_in[8];
  const float* dbq = (const float*)d_in[9];
  const float* dwk = (const float*)d_in[10];
  const float* dbk = (const float*)d_in[11];
  const float* dwv = (const float*)d_in[12];
  const float* dbv = (const float*)d_in[13];
  const float* wup = (const float*)d_in[14];
  const float* bup = (const float*)d_in[15];
  const float* wc  = (const float*)d_in[16];
  const float* bc  = (const float*)d_in[17];
  const float* w1  = (const float*)d_in[18];
  const float* b1  = (const float*)d_in[19];
  const float* w2  = (const float*)d_in[20];
  const float* b2  = (const float*)d_in[21];
  const float* g1  = (const float*)d_in[22];
  const float* be1 = (const float*)d_in[23];
  const float* g2  = (const float*)d_in[24];
  const float* be2 = (const float*)d_in[25];
  float* out = (float*)d_out;

  const size_t BS = (size_t)CB * CS;    // 16384 rows

  char* ws = (char*)d_ws;
  size_t off = 0;
  auto take = [&](size_t bytes) -> void* {
    void* p = ws + off;
    off += (bytes + 255) & ~(size_t)255;
    return p;
  };

  f16* wq16t  = (f16*)take((size_t)CDM * CDM * 2);
  f16* wk16t  = (f16*)take((size_t)CDM * CDM * 2);
  f16* wv16t  = (f16*)take((size_t)CDM * CDM * 2);
  f16* wc16t  = (f16*)take((size_t)CDM * CDM * 2);
  f16* w116t  = (f16*)take((size_t)CDM * CDF * 2);
  f16* w216t  = (f16*)take((size_t)CDM * CDF * 2);
  f16* wup16t = (f16*)take((size_t)CDD * CDD * 2);
  f16* xe16   = (f16*)take(BS * CDM * 2);
  float* q32  = (float*)take(BS * CDM * 4);
  float* k32  = (float*)take(BS * CDM * 4);
  float* v32  = (float*)take(BS * CDM * 4);
  f16* qc16   = (f16*)take(BS * CDM * 2);
  f16* kc16   = (f16*)take(BS * CDM * 2);
  f16* vc16   = (f16*)take(BS * CDM * 2);
  size_t pooled = (size_t)CB * CH * CL * CDD;
  f16* qp16   = (f16*)take(pooled * 2);
  f16* kp16   = (f16*)take(pooled * 2);
  f16* vpT16  = (f16*)take(pooled * 2);
  f16* o16    = (f16*)take(pooled * 2);
  f16* ou16   = (f16*)take(BS * CDM * 2);
  float* attn32 = (float*)take(BS * CDM * 4);
  float* x1_32  = (float*)take(BS * CDM * 4);
  f16* x1_16    = (f16*)take(BS * CDM * 2);
  f16* h1_16    = (f16*)take(BS * CDF * 2);
  float* ffw32  = (float*)take(BS * CDM * 4);

  dim3 blk(256);

  // 1) weight prep: f32 [K,N] -> f16 [N,K]
  transpose_f32_f16<<<dim3(CDM/32, CDM/32), blk, 0, stream>>>(wq, wq16t, CDM, CDM);
  transpose_f32_f16<<<dim3(CDM/32, CDM/32), blk, 0, stream>>>(wk, wk16t, CDM, CDM);
  transpose_f32_f16<<<dim3(CDM/32, CDM/32), blk, 0, stream>>>(wv, wv16t, CDM, CDM);
  transpose_f32_f16<<<dim3(CDM/32, CDM/32), blk, 0, stream>>>(wc, wc16t, CDM, CDM);
  transpose_f32_f16<<<dim3(CDF/32, CDM/32), blk, 0, stream>>>(w1, w116t, CDM, CDF);
  transpose_f32_f16<<<dim3(CDM/32, CDF/32), blk, 0, stream>>>(w2, w216t, CDF, CDM);
  transpose_f32_f16<<<dim3(CDD/32, CDD/32), blk, 0, stream>>>(wup, wup16t, CDD, CDD);

  // 2) x_embed (f16 GEMM input)
  add_embed<<<dim3((unsigned)(BS * CDM / 256)), blk, 0, stream>>>(x_enc, x_pos, xe16);

  // 3) QKV projections (scale folds the DD^-0.25 into q/k)
  dim3 ggrid(BS / BM, CDM / BN);
  gemm_wmma<<<ggrid, blk, 0, stream>>>(xe16, wq16t, bq, q32, nullptr,
                                       (int)BS, CDM, CDM, CNORM, 0);
  gemm_wmma<<<ggrid, blk, 0, stream>>>(xe16, wk16t, bk, k32, nullptr,
                                       (int)BS, CDM, CDM, CNORM, 0);
  gemm_wmma<<<ggrid, blk, 0, stream>>>(xe16, wv16t, bv, v32, nullptr,
                                       (int)BS, CDM, CDM, 1.0f, 0);

  // 4) causal depthwise conv + head split
  unsigned nElem = (unsigned)(BS * CDM / 256);
  dwconv_split<<<dim3(nElem), blk, 0, stream>>>(q32, dwq, dbq, qc16);
  dwconv_split<<<dim3(nElem), blk, 0, stream>>>(k32, dwk, dbk, kc16);
  dwconv_split<<<dim3(nElem), blk, 0, stream>>>(v32, dwv, dbv, vc16);

  // 5) seq avg-pool (V stored transposed for the P*V WMMA B operand)
  unsigned nPool = (unsigned)(pooled / 256);
  avgpool_seq<<<dim3(nPool), blk, 0, stream>>>(qc16, qp16, 0);
  avgpool_seq<<<dim3(nPool), blk, 0, stream>>>(kc16, kp16, 0);
  avgpool_seq<<<dim3(nPool), blk, 0, stream>>>(vc16, vpT16, 1);

  // 6) pooled causal flash attention
  attn_pooled<<<dim3(CL / 128, CB * CH), blk, 0, stream>>>(qp16, kp16, vpT16, o16);

  // 7) repeat-upsample + wup + head merge
  upsample_wup<<<dim3(CS / 128, CB * CH), blk, 0, stream>>>(o16, wup16t, bup, ou16);

  // 8) output projection wc
  gemm_wmma<<<ggrid, blk, 0, stream>>>(ou16, wc16t, bc, attn32, nullptr,
                                       (int)BS, CDM, CDM, 1.0f, 0);

  // 9) x1 = x_embed + LN(attn)
  residual_ln<<<dim3((unsigned)BS), blk, 0, stream>>>(x_enc, x_pos, attn32,
                                                      g1, be1, x1_32, x1_16);

  // 10) FFN: relu^2(x1@w1+b1) @ w2 + b2
  gemm_wmma<<<dim3(BS / BM, CDF / BN), blk, 0, stream>>>(
      x1_16, w116t, b1, nullptr, h1_16, (int)BS, CDF, CDM, 1.0f, 1);
  gemm_wmma<<<dim3(BS / BM, CDM / BN), blk, 0, stream>>>(
      h1_16, w216t, b2, ffw32, nullptr, (int)BS, CDM, CDF, 1.0f, 0);

  // 11) out = x1 + LN(ffw)
  residual_ln<<<dim3((unsigned)BS), blk, 0, stream>>>(x1_32, nullptr, ffw32,
                                                      g2, be2, out, nullptr);
}
